// SelectiveSSM_53343493816406
// MI455X (gfx1250) — matrix operand order repaired and checked by reading, not compiled, not run on hardware
//
#include <hip/hip_runtime.h>
#include <hip/hip_bf16.h>

#define BATCH   2
#define SEQ     2048
#define DMODEL  1024
#define DINNER  2048
#define DSTATE  16
#define DTRANK  64
#define DCONV   4
#define NROWS   (BATCH * SEQ)      // 4096 token rows

typedef unsigned short u16;
typedef __attribute__((ext_vector_type(16))) __bf16 v16bf;
typedef __attribute__((ext_vector_type(8)))  float  v8f;
typedef unsigned int uint32x4v __attribute__((ext_vector_type(4)));
typedef int          int32x4v  __attribute__((ext_vector_type(4)));
typedef int          int32x8v  __attribute__((ext_vector_type(8)));

struct __align__(16) U4 { unsigned a[4]; };   // POD 16B vector for b128 LDS reads
union Frag16 { v16bf v; U4 q[2]; };           // 32B WMMA operand fragment

__device__ __forceinline__ u16 f2bf(float f) {
    unsigned u = __float_as_uint(f);
    u += 0x7FFFu + ((u >> 16) & 1u);          // round-to-nearest-even
    return (u16)(u >> 16);
}

// ---------------------------------------------------------------- casts
__global__ void k_cast_bf16(const float* __restrict__ src, u16* __restrict__ dst, int n) {
    int i = blockIdx.x * blockDim.x + threadIdx.x;
    if (i < n) dst[i] = f2bf(src[i]);
}

// x_proj_w [96,2048] f32 -> [128,2048] bf16, rows 96..127 zeroed (pad for WMMA tiling)
__global__ void k_cast_pad_xproj(const float* __restrict__ src, u16* __restrict__ dst) {
    int i = blockIdx.x * blockDim.x + threadIdx.x;
    if (i >= 128 * DINNER) return;
    int r = i / DINNER;
    dst[i] = (r < (DTRANK + 2 * DSTATE)) ? f2bf(src[i]) : (u16)0;
}

// dt_low: x_dbl f32 [4096,128] -> bf16 [4096,64]
__global__ void k_extract_dtlow(const float* __restrict__ xdbl, u16* __restrict__ dst) {
    int i = blockIdx.x * blockDim.x + threadIdx.x;
    if (i >= NROWS * DTRANK) return;
    int r = i >> 6, c = i & 63;
    dst[i] = f2bf(xdbl[r * 128 + c]);
}

// ------------------------------------------------- TDM: async 2-D tile -> LDS
// Loads a (tileRows x 32) bf16 tile (row stride = ld elements) into LDS at ldsOff.
// D# layout per CDNA5 ISA ch.8: group0 = {count,lds,gaddr,type}, group1 = dims/strides.
__device__ __forceinline__ void tdm_load_tile(const u16* g, unsigned ldsOff,
                                              int tileRows, int ld, int dim0, int dim1) {
    unsigned long long ga = (unsigned long long)(size_t)g;
    uint32x4v g0;
    g0[0] = 1u;                                              // count=1 (valid), no gather
    g0[1] = ldsOff;                                          // LDS byte address
    g0[2] = (unsigned)ga;                                    // global_addr[31:0]
    g0[3] = ((unsigned)(ga >> 32) & 0x01FFFFFFu) | 0x80000000u; // addr[56:32] | type=2
    int32x8v g1;
    g1[0] = (int)(1u << 16);                                 // data_size=1 (2 bytes)
    g1[1] = (int)(((unsigned)dim0 & 0xFFFFu) << 16);         // tensor_dim0 lo16
    g1[2] = (int)(((unsigned)dim0 >> 16) | (((unsigned)dim1 & 0xFFFFu) << 16));
    g1[3] = (int)(((unsigned)dim1 >> 16) | (32u << 16));     // tile_dim0 = 32 (K)
    g1[4] = (int)((unsigned)tileRows & 0xFFFFu);             // tile_dim1; tile_dim2=0
    g1[5] = ld;                                              // tensor_dim0_stride lo32
    g1[6] = 0;
    g1[7] = 0;
    int32x4v z4; z4[0] = z4[1] = z4[2] = z4[3] = 0;          // 2-D: groups 2/3 unused
#if defined(__clang_major__) && __clang_major__ >= 23
    int32x8v z8;
    #pragma unroll
    for (int i = 0; i < 8; ++i) z8[i] = 0;
    __builtin_amdgcn_tensor_load_to_lds(g0, g1, z4, z4, z8, 0);
#else
    __builtin_amdgcn_tensor_load_to_lds(g0, g1, z4, z4, 0);
#endif
}

// ---------------------------------------------------------------- bf16 WMMA GEMM
// C[M,N] (f32) = A[M,K] (bf16,row-major,lda) * W[N,K]^T (bf16,row-major,ldb)
// Block tile 128x64, BK=32, 8 waves of 32x32. TDM-fed, double-buffered LDS.
// Requires M%128==0, N%64==0, K%32==0.
#define STAGE_BYTES 12288              // A tile 8KB + B tile 4KB
#define STAGE_ELEMS 6144
__global__ __launch_bounds__(256)
void k_gemm_bf16(const u16* __restrict__ A, const u16* __restrict__ W,
                 float* __restrict__ C, int K, int lda, int ldb, int ldc) {
    // Single LDS symbol -> allocated at offset 0; TDM descriptors use raw offsets.
    __shared__ __align__(16) u16 Smem[2 * STAGE_ELEMS];      // 24 KB double buffer

    const int tid   = threadIdx.x;
    const int lane  = tid & 31;
    const int wid   = tid >> 5;
    const int waveM = (wid & 3) * 32;
    const int waveN = (wid >> 2) * 32;
    const int lm    = lane & 15;
    const int aK    = (lane < 16) ? 0 : 8;    // A frag: K 0-7/16-23 vs 8-15/24-31
    const int bK    = (lane < 16) ? 0 : 8;    // B frag: K 0-7/16-23 vs 8-15/24-31

    const int rowsA = gridDim.y * 128;
    const int rowsB = gridDim.x * 64;
    const u16* Abase = A + (size_t)blockIdx.y * 128 * lda;
    const u16* Wbase = W + (size_t)blockIdx.x * 64 * ldb;

    v8f acc[2][2];
    #pragma unroll
    for (int mi = 0; mi < 2; ++mi)
        #pragma unroll
        for (int ni = 0; ni < 2; ++ni)
            #pragma unroll
            for (int r = 0; r < 8; ++r)
                acc[mi][ni][r] = 0.0f;

    // Prologue: TDM stage 0 into buffer 0 (wave 0 only; EXEC-independent DMA).
    if (wid == 0) {
        tdm_load_tile(Abase, 0u, 128, lda, K, rowsA);
        tdm_load_tile(Wbase, 8192u, 64, ldb, K, rowsB);
    }

    int s = 0;
    for (int k = 0; k < K; k += 32, s ^= 1) {
        if (wid == 0) {
            if (k + 32 < K) {
                // Kick DMA for next stage into the other buffer, then wait until
                // only those 2 transfers remain in flight (in-order TENSORcnt).
                unsigned nb = (unsigned)(s ^ 1) * STAGE_BYTES;
                tdm_load_tile(Abase + (k + 32), nb, 128, lda, K, rowsA);
                tdm_load_tile(Wbase + (k + 32), nb + 8192u, 64, ldb, K, rowsB);
                __builtin_amdgcn_s_wait_tensorcnt(2);
            } else {
                __builtin_amdgcn_s_wait_tensorcnt(0);
            }
        }
        __syncthreads();                       // publish TDM-written LDS stage s

        // TDM writes LDS behind the compiler's back (descriptor holds a raw
        // offset). Escape the LDS pointer into an asm with a memory clobber
        // (INPUT-only, so the pointer SSA value keeps its addrspace(3)
        // provenance and the loads stay ds_load_b128, not flat_load).
        const u16* As = Smem + s * STAGE_ELEMS;             // 128 x 32
        asm volatile("" : : "s"(As) : "memory");
        const u16* Bs = As + 4096;                          // 64 x 32

        Frag16 fa[2], fb[2];
        #pragma unroll
        for (int mi = 0; mi < 2; ++mi) {
            const U4* p = reinterpret_cast<const U4*>(&As[(waveM + mi * 16 + lm) * 32 + aK]);
            fa[mi].q[0] = p[0];        // K aK..aK+7
            fa[mi].q[1] = p[2];        // K aK+16..aK+23
        }
        #pragma unroll
        for (int ni = 0; ni < 2; ++ni) {
            const U4* p = reinterpret_cast<const U4*>(&Bs[(waveN + ni * 16 + lm) * 32 + bK]);
            fb[ni].q[0] = p[0];        // K bK..bK+7
            fb[ni].q[1] = p[2];        // K bK+16..bK+23
        }

        #pragma unroll
        for (int mi = 0; mi < 2; ++mi)
            #pragma unroll
            for (int ni = 0; ni < 2; ++ni)
                acc[mi][ni] = __builtin_amdgcn_wmma_f32_16x16x32_bf16(
                    false, fa[mi].v, false, fb[ni].v, (short)0, acc[mi][ni], false, false);

        __syncthreads();                       // stage s reads done before re-fill
    }

    const int cn    = blockIdx.x * 64 + waveN + (lane & 15);
    const int rbase = blockIdx.y * 128 + waveM + ((lane >> 4) << 3);
    #pragma unroll
    for (int mi = 0; mi < 2; ++mi)
        #pragma unroll
        for (int ni = 0; ni < 2; ++ni)
            #pragma unroll
            for (int r = 0; r < 8; ++r)
                C[(size_t)(rbase + mi * 16 + r) * ldc + cn + ni * 16] = acc[mi][ni][r];
}

// ------------------------------------------------- causal depthwise conv + SiLU
__global__ void k_conv_silu(const float* __restrict__ xz,
                            const float* __restrict__ conv_w,
                            const float* __restrict__ conv_b,
                            float* __restrict__ xc_f32,
                            u16* __restrict__ xc_bf16) {
    int i = blockIdx.x * blockDim.x + threadIdx.x;
    if (i >= NROWS * DINNER) return;
    int d   = i & (DINNER - 1);
    int row = i >> 11;                  // b*SEQ + l
    int l   = row & (SEQ - 1);
    float acc = conv_b[d];
    #pragma unroll
    for (int j = 0; j < DCONV; ++j) {
        int lp = l - (DCONV - 1) + j;
        if (lp >= 0)
            acc += conv_w[d * DCONV + j] *
                   xz[(size_t)(row - (DCONV - 1) + j) * (2 * DINNER) + d];
    }
    float s = acc / (1.0f + __expf(-acc));     // silu
    xc_f32[i]  = s;
    xc_bf16[i] = f2bf(s);
}

// ------------------------------------------------- dt = softplus(dt_raw + bias)
__global__ void k_softplus(float* __restrict__ dt, const float* __restrict__ bias) {
    int i = blockIdx.x * blockDim.x + threadIdx.x;
    if (i >= NROWS * DINNER) return;
    float v = dt[i] + bias[i & (DINNER - 1)];
    dt[i] = (v > 20.0f) ? v : log1pf(__expf(v));
}

// ------------------------------------------------- selective scan (state-parallel)
// lane = (b,d,n); 16 lanes per channel, h recurrence per lane, shfl_xor tree for y=h.C
__global__ __launch_bounds__(256)
void k_scan(const float* __restrict__ dt, const float* __restrict__ xc,
            const float* __restrict__ xdbl, const float* __restrict__ xz,
            const float* __restrict__ A_log, const float* __restrict__ Dvec,
            u16* __restrict__ out_bf16) {
    int tid = blockIdx.x * blockDim.x + threadIdx.x;   // 65536 lanes
    int n = tid & 15;
    int d = (tid >> 4) & (DINNER - 1);
    int b = tid >> 15;
    float An = -__expf(A_log[d * DSTATE + n]);
    float Dd = Dvec[d];
    float h = 0.0f;
    size_t rowBase = (size_t)b * SEQ;
    for (int t = 0; t < SEQ; ++t) {
        size_t row = rowBase + t;
        float dtv = dt[row * DINNER + d];              // broadcast over 16 lanes
        float xv  = xc[row * DINNER + d];
        float Bn  = xdbl[row * 128 + DTRANK + n];
        float Cn  = xdbl[row * 128 + DTRANK + DSTATE + n];
        h = __expf(dtv * An) * h + dtv * Bn * xv;
        float y = h * Cn;
        y += __shfl_xor(y, 1);
        y += __shfl_xor(y, 2);
        y += __shfl_xor(y, 4);
        y += __shfl_xor(y, 8);
        if (n == 0) {
            float z = xz[row * (2 * DINNER) + DINNER + d];
            float o = (y + xv * Dd) * (z / (1.0f + __expf(-z)));
            out_bf16[row * DINNER + d] = f2bf(o);
        }
    }
}

// ---------------------------------------------------------------- launcher
extern "C" void kernel_launch(void* const* d_in, const int* in_sizes, int n_in,
                              void* d_out, int out_size, void* d_ws, size_t ws_size,
                              hipStream_t stream) {
    const float* x         = (const float*)d_in[0];
    const float* in_proj_w = (const float*)d_in[1];
    const float* conv_w    = (const float*)d_in[2];
    const float* conv_b    = (const float*)d_in[3];
    const float* xproj_w   = (const float*)d_in[4];
    const float* dtproj_w  = (const float*)d_in[5];
    const float* dtproj_b  = (const float*)d_in[6];
    const float* A_log     = (const float*)d_in[7];
    const float* Dvec      = (const float*)d_in[8];
    const float* outproj_w = (const float*)d_in[9];
    float* out = (float*)d_out;

    char* ws = (char*)d_ws;
    size_t off = 0;
    auto take = [&](size_t bytes) -> char* {
        char* p = ws + off;
        off += (bytes + 255) & ~(size_t)255;
        return p;
    };
    float* xz     = (float*)take((size_t)NROWS * 4096 * 4);     // [4096,4096]
    float* xc_f32 = (float*)take((size_t)NROWS * DINNER * 4);   // [4096,2048]
    float* dt_f32 = (float*)take((size_t)NROWS * DINNER * 4);   // [4096,2048]
    float* xdbl   = (float*)take((size_t)NROWS * 128 * 4);      // [4096,128] (96 used)
    u16* x_bf     = (u16*)take((size_t)NROWS * DMODEL * 2);
    u16* w_in_bf  = (u16*)take((size_t)4096 * DMODEL * 2);
    u16* xc_bf    = (u16*)take((size_t)NROWS * DINNER * 2);
    u16* w_xp_bf  = (u16*)take((size_t)128 * DINNER * 2);
    u16* dtlow_bf = (u16*)take((size_t)NROWS * DTRANK * 2);
    u16* w_dt_bf  = (u16*)take((size_t)DINNER * DTRANK * 2);
    u16* out_bf   = (u16*)take((size_t)NROWS * DINNER * 2);
    u16* w_out_bf = (u16*)take((size_t)DMODEL * DINNER * 2);

    dim3 blk(256);
    k_cast_bf16<<<(NROWS * DMODEL + 255) / 256, blk, 0, stream>>>(x, x_bf, NROWS * DMODEL);
    k_cast_bf16<<<(4096 * DMODEL + 255) / 256, blk, 0, stream>>>(in_proj_w, w_in_bf, 4096 * DMODEL);
    k_cast_pad_xproj<<<(128 * DINNER + 255) / 256, blk, 0, stream>>>(xproj_w, w_xp_bf);
    k_cast_bf16<<<(DINNER * DTRANK + 255) / 256, blk, 0, stream>>>(dtproj_w, w_dt_bf, DINNER * DTRANK);
    k_cast_bf16<<<(DMODEL * DINNER + 255) / 256, blk, 0, stream>>>(outproj_w, w_out_bf, DMODEL * DINNER);

    // xz = x @ in_proj_w^T : [4096,1024] x [4096,1024]^T -> [4096,4096]
    k_gemm_bf16<<<dim3(4096 / 64, NROWS / 128), blk, 0, stream>>>(
        x_bf, w_in_bf, xz, DMODEL, DMODEL, DMODEL, 4096);

    k_conv_silu<<<(NROWS * DINNER + 255) / 256, blk, 0, stream>>>(
        xz, conv_w, conv_b, xc_f32, xc_bf);

    // x_dbl = xc @ x_proj_w^T : [4096,2048] x [128,2048]^T -> [4096,128]
    k_gemm_bf16<<<dim3(128 / 64, NROWS / 128), blk, 0, stream>>>(
        xc_bf, w_xp_bf, xdbl, DINNER, DINNER, DINNER, 128);

    k_extract_dtlow<<<(NROWS * DTRANK + 255) / 256, blk, 0, stream>>>(xdbl, dtlow_bf);

    // dt_raw = dt_low @ dt_proj_w^T : [4096,64] x [2048,64]^T -> [4096,2048]
    k_gemm_bf16<<<dim3(DINNER / 64, NROWS / 128), blk, 0, stream>>>(
        dtlow_bf, w_dt_bf, dt_f32, DTRANK, DTRANK, DTRANK, DINNER);

    k_softplus<<<(NROWS * DINNER + 255) / 256, blk, 0, stream>>>(dt_f32, dtproj_b);

    k_scan<<<(BATCH * DINNER * DSTATE) / 256, blk, 0, stream>>>(
        dt_f32, xc_f32, xdbl, xz, A_log, Dvec, out_bf);

    // out = ssm_out @ out_proj_w^T : [4096,2048] x [1024,2048]^T -> [4096,1024]
    k_gemm_bf16<<<dim3(DMODEL / 64, NROWS / 128), blk, 0, stream>>>(
        out_bf, w_out_bf, out, DINNER, DINNER, DINNER, DMODEL);
}
